// PinSage_24481313587345
// MI455X (gfx1250) — compile-verified, matrix operand order checked by reading, not run on hardware
//
#include <hip/hip_runtime.h>

typedef __attribute__((ext_vector_type(16))) _Float16 v16h;
typedef __attribute__((ext_vector_type(8)))  float    v8f;

#define F_DIM 256
#define TNB   16
#define LSLOPE 0.01f

// ---------------------------------------------------------------------------
// helpers
// ---------------------------------------------------------------------------
__device__ __forceinline__ v8f wmma_f16(v16h a, v16h b, v8f c) {
  // D = A x B + C, 16x16x32, f16 in / f32 accumulate (CDNA5 wave32 WMMA)
  return __builtin_amdgcn_wmma_f32_16x16x32_f16(
      /*neg_a=*/false, a, /*neg_b=*/false, b,
      /*c_mod=*/(short)0, c, /*reuse_a=*/false, /*reuse_b=*/false);
}

// Load one lane's 16-half WMMA operand fragment from a row-major f16 row.
// Fragment layout (ISA 7.12.2, 16-bit A/B): lane half h -> K {kb..kb+7, kb+16..kb+23}
__device__ __forceinline__ v16h load_frag(const _Float16* __restrict__ base, int kb) {
  v16h a;
  ((uint4*)&a)[0] = *(const uint4*)(base + kb);
  ((uint4*)&a)[1] = *(const uint4*)(base + kb + 16);
  return a;
}

// resolve a node row: if map[g] >= 0, row lives in repl (layer-0 output)
__device__ __forceinline__ const float* resolve_row(
    const float* __restrict__ h, const int* __restrict__ map,
    const float* __restrict__ repl, int g) {
  if (map != nullptr) {
    int p = map[g];
    if (p >= 0) return repl + (size_t)p * F_DIM;
  }
  return h + (size_t)g * F_DIM;
}

// ---------------------------------------------------------------------------
// small utility kernels
// ---------------------------------------------------------------------------
__global__ void f32_to_f16_kernel(const float* __restrict__ src,
                                  _Float16* __restrict__ dst, int n) {
  int i = blockIdx.x * blockDim.x + threadIdx.x;
  if (i < n) dst[i] = (_Float16)src[i];
}

__global__ void fill_i32_kernel(int* __restrict__ p, int v, int n) {
  int i = blockIdx.x * blockDim.x + threadIdx.x;
  if (i < n) p[i] = v;
}

__global__ void build_map_kernel(const int* __restrict__ nodeset,
                                 int* __restrict__ map, int n) {
  int i = blockIdx.x * blockDim.x + threadIdx.x;
  if (i < n) atomicMax(&map[nodeset[i]], i);  // "last index wins" deterministically
}

__global__ void final_gather_kernel(const float* __restrict__ new1,
                                    const int* __restrict__ map1,
                                    const int* __restrict__ ns1,
                                    float* __restrict__ out) {
  int i = blockIdx.x;                 // minibatch row
  int pos = map1[ns1[i]];             // winner position (>= 0 always)
  out[(size_t)i * F_DIM + threadIdx.x] = new1[(size_t)pos * F_DIM + threadIdx.x];
}

// ---------------------------------------------------------------------------
// PinSAGE conv: one workgroup (16 wave32, 512 threads) handles 16 output nodes.
// Wave w owns output N-tile w (cols 16w..16w+15).
// Stage 1: Z = lrelu(h_nb @ Q^T + Qb), weighted mean over T=16  -> h_agg
//   8 M-blocks of 32 gathered rows; next block's HBM gather is issued into
//   registers before the WMMA phase so load latency hides under compute.
// Stage 2: h_new = lrelu([h_self | h_agg] @ W^T + Wb), L2-normalize rows
// ---------------------------------------------------------------------------
__global__ __launch_bounds__(512) void pinsage_conv_kernel(
    const float* __restrict__ h,
    const int* __restrict__ map, const float* __restrict__ repl,
    const _Float16* __restrict__ Qh, const float* __restrict__ Qb,
    const _Float16* __restrict__ Wh, const float* __restrict__ Wb,
    const float* __restrict__ nbw, const int* __restrict__ nodeset,
    const int* __restrict__ nbn, float* __restrict__ outbuf) {
  __shared__ _Float16 A_blk[32][264];   // 32 gathered neighbor rows (f16), padded
  __shared__ _Float16 Hcat[16][520];    // [self(256) | agg(256)] f16, padded
  __shared__ float    Hagg[16][264];    // stage-1 agg (f32); reused as Hout in stage 2
  __shared__ float    wl[16][16];       // neighbor weights
  __shared__ float    denomv[16];
  __shared__ float    norms[16];

  const int tid  = threadIdx.x;
  const int lane = tid & 31;
  const int wv   = tid >> 5;         // wave 0..15 -> owns N-tile wv
  const int hl   = lane >> 4;        // lane half (fragment K-split)
  const int l15  = lane & 15;
  const long node_base = (long)blockIdx.x * 16;

  // gather-staging mapping: 16 threads per row, 16 floats (4x float4) each
  const int arow_id = tid >> 4;      // 0..31
  const int aseg    = tid & 15;

  // ---- prologue ----------------------------------------------------------
  if (tid < 256) {                   // neighbor weights
    int n = tid >> 4, t = tid & 15;
    wl[n][t] = nbw[(node_base + n) * TNB + t];
  }
  {                                  // self rows f32 -> f16 into Hcat[:, :256]
    int row = tid >> 5, seg = tid & 31;          // 32 threads/row, 8 floats each
    int g = nodeset[node_base + row];
    const float* src = resolve_row(h, map, repl, g);
    _Float16* dst = &Hcat[row][seg * 8];
#pragma unroll
    for (int j = 0; j < 2; ++j) {
      float4 v = ((const float4*)src)[seg * 2 + j];
      union { _Float16 hh[4]; uint2 u; } pk;
      pk.hh[0] = (_Float16)v.x; pk.hh[1] = (_Float16)v.y;
      pk.hh[2] = (_Float16)v.z; pk.hh[3] = (_Float16)v.w;
      *(uint2*)(dst + j * 4) = pk.u;
    }
  }
  // issue gather for M-block 0 (stays in flight; waited at first use)
  float4 stg[4];
  {
    int nd = arow_id >> 4, t = arow_id & 15;
    int g = nbn[(node_base + nd) * TNB + t];
    const float* src = resolve_row(h, map, repl, g);
#pragma unroll
    for (int j = 0; j < 4; ++j) stg[j] = ((const float4*)src)[aseg * 4 + j];
  }
  __syncthreads();
  if (tid < 16) {
    float s = 0.f;
#pragma unroll
    for (int t = 0; t < TNB; ++t) s += wl[tid][t];
    denomv[tid] = (s == 0.f) ? 1.f : s;           // safediv
    norms[tid] = 0.f;
  }

  // ---- stage 1: 8 software-pipelined M-blocks (32 rows = 2 nodes x 16 nbrs)
#pragma unroll 1
  for (int b = 0; b < 8; ++b) {
    // commit staged rows (waits the in-flight loads) -> f16 A_blk
    {
      _Float16* dst = &A_blk[arow_id][aseg * 16];
#pragma unroll
      for (int j = 0; j < 4; ++j) {
        union { _Float16 hh[4]; uint2 u; } pk;
        pk.hh[0] = (_Float16)stg[j].x; pk.hh[1] = (_Float16)stg[j].y;
        pk.hh[2] = (_Float16)stg[j].z; pk.hh[3] = (_Float16)stg[j].w;
        *(uint2*)(dst + j * 4) = pk.u;
      }
    }
    __syncthreads();
    // issue gather for block b+1 (overlaps with WMMA phase below)
    if (b < 7) {
      int nd = 2 * (b + 1) + (arow_id >> 4), t = arow_id & 15;
      int g = nbn[(node_base + nd) * TNB + t];
      const float* src = resolve_row(h, map, repl, g);
#pragma unroll
      for (int j = 0; j < 4; ++j) stg[j] = ((const float4*)src)[aseg * 4 + j];
    }

    // WMMA phase: 2 M-tiles x 1 N-tile per wave, K = 256 (8 K-steps)
    v8f acc0 = {}, acc1 = {};
    const _Float16* brow  = Qh + (size_t)(wv * 16 + l15) * F_DIM;  // B col = Qw row
    const _Float16* arow0 = &A_blk[l15][0];
    const _Float16* arow1 = &A_blk[16 + l15][0];
#pragma unroll
    for (int ks = 0; ks < 8; ++ks) {
      int kb = ks * 32 + hl * 8;
      v16h bf = load_frag(brow, kb);    // loop-invariant across b: hoisted to regs
      v16h a0 = load_frag(arow0, kb);
      v16h a1 = load_frag(arow1, kb);
      acc0 = wmma_f16(a0, bf, acc0);
      acc1 = wmma_f16(a1, bf, acc1);
    }

    // epilogue: bias + leaky-relu + weighted mean over the tile's M (= T) dim
    auto epi1 = [&](v8f acc, int nd) {
      int col = wv * 16 + l15;
      float bias = Qb[col];
      float partial = 0.f;
#pragma unroll
      for (int r = 0; r < 8; ++r) {
        int t = r + 8 * hl;
        float z = acc[r] + bias;
        z = (z > 0.f) ? z : LSLOPE * z;
        partial += wl[nd][t] * z;
      }
      partial += __shfl_xor(partial, 16, 32);     // combine lane halves (M 0-7 | 8-15)
      if (lane < 16) Hagg[nd][col] = partial / denomv[nd];
    };
    epi1(acc0, 2 * b + 0);
    epi1(acc1, 2 * b + 1);
    __syncthreads();   // everyone done with A_blk before next commit
  }

  // ---- convert h_agg (f32) -> Hcat[:, 256:512] (f16) ----------------------
  {
    int n = tid >> 5, seg = tid & 31;
#pragma unroll
    for (int j = 0; j < 8; ++j)
      Hcat[n][256 + seg * 8 + j] = (_Float16)Hagg[n][seg * 8 + j];
  }
  __syncthreads();

  // ---- stage 2: [16 x 512] @ W^T -> [16 x 256], K = 512 (16 K-steps) ------
  v8f acc2 = {};
  {
    const _Float16* wrow = Wh + (size_t)(wv * 16 + l15) * (2 * F_DIM);
    const _Float16* arow = &Hcat[l15][0];
#pragma unroll
    for (int ks = 0; ks < 16; ++ks) {
      int kb = ks * 32 + hl * 8;
      v16h a  = load_frag(arow, kb);
      v16h bf = load_frag(wrow, kb);
      acc2 = wmma_f16(a, bf, acc2);
    }
  }
  // epilogue: bias + leaky-relu; store to Hout (aliases Hagg) + norm reduce
  {
    int col = wv * 16 + l15;
    float bias = Wb[col];
#pragma unroll
    for (int r = 0; r < 8; ++r) {
      int nd = r + 8 * hl;
      float z = acc2[r] + bias;
      z = (z > 0.f) ? z : LSLOPE * z;
      Hagg[nd][col] = z;
      atomicAdd(&norms[nd], z * z);
    }
  }
  __syncthreads();

  // ---- normalize rows and store ------------------------------------------
  {
    int nd = tid >> 5, seg = tid & 31;
    float nrm = sqrtf(norms[nd]);
    if (nrm == 0.f) nrm = 1.f;                    // safediv
    float* orow = outbuf + (size_t)(node_base + nd) * F_DIM;
#pragma unroll
    for (int j = 0; j < 2; ++j) {
      float4 v = *(const float4*)(&Hagg[nd][seg * 8 + j * 4]);
      v.x /= nrm; v.y /= nrm; v.z /= nrm; v.w /= nrm;
      *(float4*)(orow + seg * 8 + j * 4) = v;
    }
  }
}

// ---------------------------------------------------------------------------
// launcher
// ---------------------------------------------------------------------------
extern "C" void kernel_launch(void* const* d_in, const int* in_sizes, int n_in,
                              void* d_out, int out_size, void* d_ws, size_t ws_size,
                              hipStream_t stream) {
  (void)n_in; (void)out_size; (void)ws_size;
  const float* h    = (const float*)d_in[0];
  const float* Q0w  = (const float*)d_in[1];
  const float* Q0b  = (const float*)d_in[2];
  const float* W0w  = (const float*)d_in[3];
  const float* W0b  = (const float*)d_in[4];
  const float* Q1w  = (const float*)d_in[5];
  const float* Q1b  = (const float*)d_in[6];
  const float* W1w  = (const float*)d_in[7];
  const float* W1b  = (const float*)d_in[8];
  const float* nbw0 = (const float*)d_in[9];
  const float* nbw1 = (const float*)d_in[10];
  const int*   ns0  = (const int*)d_in[11];
  const int*   nbn0 = (const int*)d_in[12];
  const int*   ns1  = (const int*)d_in[13];
  const int*   nbn1 = (const int*)d_in[14];

  const int L0 = in_sizes[11];
  const int L1 = in_sizes[13];
  const int nNodes = in_sizes[0] / F_DIM;

  char* ws = (char*)d_ws;
  size_t off = 0;
  auto carve = [&](size_t bytes) -> void* {
    void* p = ws + off;
    off = (off + bytes + 255) & ~(size_t)255;
    return p;
  };
  const int nq = F_DIM * F_DIM;            // 65536
  const int nw = F_DIM * 2 * F_DIM;        // 131072
  _Float16* Qh0 = (_Float16*)carve((size_t)nq * 2);
  _Float16* Wh0 = (_Float16*)carve((size_t)nw * 2);
  _Float16* Qh1 = (_Float16*)carve((size_t)nq * 2);
  _Float16* Wh1 = (_Float16*)carve((size_t)nw * 2);
  int*   map0 = (int*)carve((size_t)nNodes * 4);
  int*   map1 = (int*)carve((size_t)nNodes * 4);
  float* new0 = (float*)carve((size_t)L0 * F_DIM * 4);
  float* new1 = (float*)carve((size_t)L1 * F_DIM * 4);

  // 1) weights -> f16
  f32_to_f16_kernel<<<(nq + 255) / 256, 256, 0, stream>>>(Q0w, Qh0, nq);
  f32_to_f16_kernel<<<(nw + 255) / 256, 256, 0, stream>>>(W0w, Wh0, nw);
  f32_to_f16_kernel<<<(nq + 255) / 256, 256, 0, stream>>>(Q1w, Qh1, nq);
  f32_to_f16_kernel<<<(nw + 255) / 256, 256, 0, stream>>>(W1w, Wh1, nw);

  // 2) scatter-winner maps (node -> last position in nodeset)
  fill_i32_kernel<<<(nNodes + 255) / 256, 256, 0, stream>>>(map0, -1, nNodes);
  fill_i32_kernel<<<(nNodes + 255) / 256, 256, 0, stream>>>(map1, -1, nNodes);
  build_map_kernel<<<(L0 + 255) / 256, 256, 0, stream>>>(ns0, map0, L0);
  build_map_kernel<<<(L1 + 255) / 256, 256, 0, stream>>>(ns1, map1, L1);

  // 3) layer 0: reads original h, writes new0 (h untouched)
  pinsage_conv_kernel<<<L0 / 16, 512, 0, stream>>>(
      h, nullptr, nullptr, Qh0, Q0b, Wh0, W0b, nbw0, ns0, nbn0, new0);

  // 4) layer 1: reads h overlaid with new0 via map0, writes new1
  pinsage_conv_kernel<<<L1 / 16, 512, 0, stream>>>(
      h, map0, new0, Qh1, Q1b, Wh1, W1b, nbw1, ns1, nbn1, new1);

  // 5) out = (h after nodeset1 scatter)[nodeset1]  (duplicate-aware)
  final_gather_kernel<<<L1, 256, 0, stream>>>(new1, map1, ns1, (float*)d_out);
}
